// PivNet_69810398429458
// MI455X (gfx1250) — compile-verified, matrix operand order checked by reading, not run on hardware
//
#include <hip/hip_runtime.h>
#include <hip/hip_bf16.h>
#include <math.h>

typedef _Float16 f16;
typedef __attribute__((ext_vector_type(16))) _Float16 v16h;
typedef __attribute__((ext_vector_type(8)))  _Float16 v8h;
typedef __attribute__((ext_vector_type(8)))  float    v8f;
typedef __attribute__((ext_vector_type(4)))  int      v4i;

#define BATCH   524288
#define GRIDN   16
#define KNN     100
#define HID     512
#define FEAT    105
#define FEATP   128   // feature K padded to WMMA K granularity
#define KOUTP   128   // output N padded to 32-col pair granularity
#define MTILE   128   // rows per block (8 waves x 16 rows)
#define LSTRIDE 520   // f16 per LDS act row: 512 + 8 pad -> conflict-free frag loads
#define BSLOT   (32 * 520)  // f16 per B staging slot: 32 N-rows, padded stride

#ifndef __has_builtin
#define __has_builtin(x) 0
#endif
#if __has_builtin(__builtin_amdgcn_global_load_async_to_lds_b128)
#define HAVE_ASYNC_LDS 1
#else
#define HAVE_ASYNC_LDS 0
#endif

#define GAS __attribute__((address_space(1)))
#define LAS __attribute__((address_space(3)))

// ---------------------------------------------------------------------------
// Weight prep: f32 row-major W[K][N]  ->  f16 transposed WT[N][K] (padded)
// ---------------------------------------------------------------------------
__global__ void pivnet_prep(const float* __restrict__ W1, const float* __restrict__ W2,
                            const float* __restrict__ W3, const float* __restrict__ W4,
                            const float* __restrict__ b4,
                            f16* __restrict__ wt1, f16* __restrict__ wt2,
                            f16* __restrict__ wt3, f16* __restrict__ wt4,
                            float* __restrict__ b4p) {
  int t = blockIdx.x * 256 + threadIdx.x;
  if (t < HID * FEATP) {                       // WT1 [512][128]
    int n = t / FEATP, k = t % FEATP;
    wt1[t] = (f16)((k < FEAT) ? W1[k * HID + n] : 0.0f);
  }
  if (t < HID * HID) {                         // WT2 / WT3 [512][512]
    int n = t / HID, k = t % HID;
    wt2[t] = (f16)W2[k * HID + n];
    wt3[t] = (f16)W3[k * HID + n];
  }
  if (t < KOUTP * HID) {                       // WT4 [128][512]
    int n = t / HID, k = t % HID;
    wt4[t] = (f16)((n < KNN) ? W4[k * KNN + n] : 0.0f);
  }
  if (t < KOUTP) b4p[t] = (t < KNN) ? b4[t] : 0.0f;
}

// ---------------------------------------------------------------------------
// Fragment loaders (ISA 7.12.2 16-bit layout: lane<16 -> K {0..7,16..23},
// lane>=16 -> K {8..15,24..31}; B mirrors A with W stored transposed)
// ---------------------------------------------------------------------------
__device__ __forceinline__ v16h load_afrag_lds(const f16* base, int lane) {
  int m = lane & 15, kh = lane >> 4;
  const f16* p = base + m * LSTRIDE + 8 * kh;
  v8h lo = *(const v8h*)(p);
  v8h hi = *(const v8h*)(p + 16);
  return __builtin_shufflevector(lo, hi, 0,1,2,3,4,5,6,7,8,9,10,11,12,13,14,15);
}

template<int KD>
__device__ __forceinline__ v16h load_bfrag_lds(const f16* tile, int kbase, int lane) {
  int n = lane & 15, kh = lane >> 4;
  const f16* p = tile + n * (KD + 8) + kbase + 8 * kh;
  v8h lo = *(const v8h*)(p);
  v8h hi = *(const v8h*)(p + 16);
  return __builtin_shufflevector(lo, hi, 0,1,2,3,4,5,6,7,8,9,10,11,12,13,14,15);
}

// ---------------------------------------------------------------------------
// Cooperative B-tile staging: 32 N-rows x KD f16 (contiguous 64*KD bytes) from
// global into a padded LDS tile.  CDNA5 async global->LDS DMA when available.
// ---------------------------------------------------------------------------
template<int KD>
__device__ __forceinline__ void stage_tile(f16* __restrict__ dst,
                                           const f16* __restrict__ src, int tid) {
  constexpr int NCH = (32 * KD) / 8;   // 16-byte chunks in the tile
  constexpr int CPR = KD / 8;          // chunks per row
  #pragma unroll
  for (int i = 0; i < NCH / 256; ++i) {
    int c = tid + i * 256;
    int row = c / CPR, j = c % CPR;
    f16*       d = dst + row * (KD + 8) + j * 8;
    const f16* s = src + c * 8;        // tile is contiguous in transposed layout
#if HAVE_ASYNC_LDS
    __builtin_amdgcn_global_load_async_to_lds_b128((GAS v4i*)s, (LAS v4i*)d, 0, 0);
#else
    *(v8h*)d = *(const v8h*)s;
#endif
  }
}

__device__ __forceinline__ void async_join() {
#if HAVE_ASYNC_LDS
# if __has_builtin(__builtin_amdgcn_s_wait_asynccnt)
  __builtin_amdgcn_s_wait_asynccnt(0);
# else
  asm volatile("s_wait_asynccnt 0" ::: "memory");
# endif
#endif
  __syncthreads();
}

// ---------------------------------------------------------------------------
// One MLP layer: per-wave A (16 rows) in VGPRs, block-shared double-buffered
// 32-column B tiles in LDS, dual accumulator chains + one-step B-fragment
// register prefetch, bias(+ReLU), f16 result back to LDS.
// ---------------------------------------------------------------------------
template<int KD, int NTP, bool RELU>
__device__ __forceinline__ void mlp_layer_lds(
    const f16* __restrict__ wtg, const float* __restrict__ bias,
    const f16* myA, f16* myC, f16* bb, int lane, int tid, int nlo, int mhi) {
  constexpr int AF = KD / 32;
  constexpr int TSTR = KD + 8;
  v16h a[AF];
  #pragma unroll
  for (int kt = 0; kt < AF; ++kt) a[kt] = load_afrag_lds(myA + kt * 32, lane);
  stage_tile<KD>(bb, wtg, tid);              // prologue: pair 0 -> slot 0
  async_join();
  #pragma unroll 1
  for (int p = 0; p < NTP; ++p) {
    f16* cur = bb + (p & 1) * BSLOT;
    f16* nxt = bb + ((p & 1) ^ 1) * BSLOT;
    if (p + 1 < NTP) stage_tile<KD>(nxt, wtg + (p + 1) * 32 * KD, tid);
    const f16* t0 = cur;                     // cols p*32    .. p*32+15
    const f16* t1 = cur + 16 * TSTR;         // cols p*32+16 .. p*32+31
    v8f acc0 = {}, acc1 = {};
    v16h b0 = load_bfrag_lds<KD>(t0, 0, lane);
    v16h b1 = load_bfrag_lds<KD>(t1, 0, lane);
    #pragma unroll
    for (int kt = 0; kt < AF; ++kt) {
      v16h c0 = b0, c1 = b1;
      if (kt + 1 < AF) {                     // prefetch next step's fragments
        b0 = load_bfrag_lds<KD>(t0, (kt + 1) * 32, lane);
        b1 = load_bfrag_lds<KD>(t1, (kt + 1) * 32, lane);
      }
      acc0 = __builtin_amdgcn_wmma_f32_16x16x32_f16(false, a[kt], false, c0,
                                                    (short)0, acc0, false, false);
      acc1 = __builtin_amdgcn_wmma_f32_16x16x32_f16(false, a[kt], false, c1,
                                                    (short)0, acc1, false, false);
    }
    float bv0 = bias[p * 32 + nlo];
    float bv1 = bias[p * 32 + 16 + nlo];
    #pragma unroll
    for (int r = 0; r < 8; ++r) {
      float h0 = acc0[r] + bv0, h1 = acc1[r] + bv1;
      if (RELU) { h0 = fmaxf(h0, 0.0f); h1 = fmaxf(h1, 0.0f); }
      myC[(r + mhi) * LSTRIDE + p * 32 + nlo]      = (f16)h0;
      myC[(r + mhi) * LSTRIDE + p * 32 + 16 + nlo] = (f16)h1;
    }
    async_join();                            // next pair staged & cur fully read
  }
}

// Final layer: f32 result straight to global (n < 100 guard).
template<int KD, int NTP>
__device__ __forceinline__ void mlp_layer_out(
    const f16* __restrict__ wtg, const float* __restrict__ bias,
    const f16* myA, float* __restrict__ outw, f16* bb, int lane, int tid,
    int nlo, int mhi) {
  constexpr int AF = KD / 32;
  constexpr int TSTR = KD + 8;
  v16h a[AF];
  #pragma unroll
  for (int kt = 0; kt < AF; ++kt) a[kt] = load_afrag_lds(myA + kt * 32, lane);
  stage_tile<KD>(bb, wtg, tid);
  async_join();
  #pragma unroll 1
  for (int p = 0; p < NTP; ++p) {
    f16* cur = bb + (p & 1) * BSLOT;
    f16* nxt = bb + ((p & 1) ^ 1) * BSLOT;
    if (p + 1 < NTP) stage_tile<KD>(nxt, wtg + (p + 1) * 32 * KD, tid);
    const f16* t0 = cur;
    const f16* t1 = cur + 16 * TSTR;
    v8f acc0 = {}, acc1 = {};
    v16h b0 = load_bfrag_lds<KD>(t0, 0, lane);
    v16h b1 = load_bfrag_lds<KD>(t1, 0, lane);
    #pragma unroll
    for (int kt = 0; kt < AF; ++kt) {
      v16h c0 = b0, c1 = b1;
      if (kt + 1 < AF) {
        b0 = load_bfrag_lds<KD>(t0, (kt + 1) * 32, lane);
        b1 = load_bfrag_lds<KD>(t1, (kt + 1) * 32, lane);
      }
      acc0 = __builtin_amdgcn_wmma_f32_16x16x32_f16(false, a[kt], false, c0,
                                                    (short)0, acc0, false, false);
      acc1 = __builtin_amdgcn_wmma_f32_16x16x32_f16(false, a[kt], false, c1,
                                                    (short)0, acc1, false, false);
    }
    int n0 = p * 32 + nlo, n1 = n0 + 16;
    float bv0 = bias[n0], bv1 = bias[n1];
    if (n0 < KNN) {
      #pragma unroll
      for (int r = 0; r < 8; ++r)
        outw[(r + mhi) * KNN + n0] = acc0[r] + bv0;
    }
    if (n1 < KNN) {
      #pragma unroll
      for (int r = 0; r < 8; ++r)
        outw[(r + mhi) * KNN + n1] = acc1[r] + bv1;
    }
    async_join();
  }
}

// ---------------------------------------------------------------------------
// Fused: gather + featurize + 4-layer MLP
// ---------------------------------------------------------------------------
extern __shared__ f16 smem[];   // [2*BSLOT B-staging][MTILE*LSTRIDE act] = 199680 B

__global__ __launch_bounds__(256, 1)
void pivnet_fused(const float* __restrict__ x,    const float* __restrict__ minv,
                  const float* __restrict__ maxv, const float* __restrict__ pivots,
                  const float* __restrict__ knnd,
                  const float* __restrict__ qmean, const float* __restrict__ qstd,
                  const float* __restrict__ kmean, const float* __restrict__ kstd,
                  const f16* __restrict__ wt1, const float* __restrict__ b1,
                  const f16* __restrict__ wt2, const float* __restrict__ b2,
                  const f16* __restrict__ wt3, const float* __restrict__ b3,
                  const f16* __restrict__ wt4, const float* __restrict__ b4p,
                  float* __restrict__ out) {
  const int tid  = threadIdx.x;
  const int lane = tid & 31;
  const int wave = tid >> 5;
  const int row0 = blockIdx.x * MTILE;
  f16* bb  = smem;                 // 2 B-staging slots
  f16* act = smem + 2 * BSLOT;     // [MTILE][LSTRIDE]

  // ---------------- phase 0: featurize (2 threads per row) ----------------
  {
    int r = tid >> 1;              // 0..127
    int half = tid & 1;
    int grow = row0 + r;
    float xv[4], bw2 = 0.0f;
    int idx = 0;
    #pragma unroll
    for (int d = 0; d < 4; ++d) {
      float mn = minv[d], mx = maxv[d];
      float v = x[grow * 4 + d];
      xv[d] = v;
      float w = (mx - mn) * (1.0f / GRIDN);
      bw2 += w * w;
      float s = (v - mn) / (mx - mn) * GRIDN;
      int b = (int)floorf(s);
      b = b < 0 ? 0 : (b > GRIDN - 1 ? GRIDN - 1 : b);
      idx = idx * GRIDN + b;       // radix base GRIDN, dim-major
    }
    f16* rowp = act + r * LSTRIDE;
    if (half == 0) {
      float d2 = 0.0f;
      #pragma unroll
      for (int d = 0; d < 4; ++d) {
        float t = xv[d] - pivots[idx * 4 + d];
        d2 += t * t;
        rowp[d] = (f16)((xv[d] - qmean[d]) / qstd[d]);
      }
      rowp[4] = (f16)(sqrtf(d2) / sqrtf(bw2));
      for (int k = 0; k < 50; ++k)
        rowp[5 + k] = (f16)((knnd[idx * KNN + k] - kmean[k]) / kstd[k]);
    } else {
      for (int k = 50; k < 100; ++k)
        rowp[5 + k] = (f16)((knnd[idx * KNN + k] - kmean[k]) / kstd[k]);
      #pragma unroll
      for (int c = FEAT; c < FEATP; ++c) rowp[c] = (f16)0.0f;
    }
  }
  __syncthreads();

  const f16* myA = act + (wave * 16) * LSTRIDE;   // wave-private 16-row slice
  f16*       myC = act + (wave * 16) * LSTRIDE;
  const int nlo = lane & 15, mhi = 8 * (lane >> 4);

  mlp_layer_lds<FEATP, 16, true>(wt1, b1, myA, myC, bb, lane, tid, nlo, mhi);
  mlp_layer_lds<HID,   16, true>(wt2, b2, myA, myC, bb, lane, tid, nlo, mhi);
  mlp_layer_lds<HID,   16, true>(wt3, b3, myA, myC, bb, lane, tid, nlo, mhi);
  mlp_layer_out<HID, KOUTP / 32>(wt4, b4p, myA, out + (row0 + wave * 16) * KNN,
                                 bb, lane, tid, nlo, mhi);
}

// ---------------------------------------------------------------------------
extern "C" void kernel_launch(void* const* d_in, const int* in_sizes, int n_in,
                              void* d_out, int out_size, void* d_ws, size_t ws_size,
                              hipStream_t stream) {
  const float* x      = (const float*)d_in[0];
  const float* minv   = (const float*)d_in[1];
  const float* maxv   = (const float*)d_in[2];
  const float* pivots = (const float*)d_in[3];
  const float* knnd   = (const float*)d_in[4];
  const float* qmean  = (const float*)d_in[5];
  const float* qstd   = (const float*)d_in[6];
  const float* kmean  = (const float*)d_in[7];
  const float* kstd   = (const float*)d_in[8];
  const float* W1 = (const float*)d_in[9];
  const float* b1 = (const float*)d_in[10];
  const float* W2 = (const float*)d_in[11];
  const float* b2 = (const float*)d_in[12];
  const float* W3 = (const float*)d_in[13];
  const float* b3 = (const float*)d_in[14];
  const float* W4 = (const float*)d_in[15];
  const float* b4 = (const float*)d_in[16];

  f16* wt1 = (f16*)d_ws;                       // [512][128]
  f16* wt2 = wt1 + HID * FEATP;                // [512][512]
  f16* wt3 = wt2 + HID * HID;                  // [512][512]
  f16* wt4 = wt3 + HID * HID;                  // [128][512]
  float* b4p = (float*)(wt4 + KOUTP * HID);    // [128]

  pivnet_prep<<<(HID * HID + 255) / 256, 256, 0, stream>>>(
      W1, W2, W3, W4, b4, wt1, wt2, wt3, wt4, b4p);

  size_t lds = (size_t)(2 * BSLOT + MTILE * LSTRIDE) * sizeof(f16);  // 199680 B
  pivnet_fused<<<BATCH / MTILE, 256, lds, stream>>>(
      x, minv, maxv, pivots, knnd, qmean, qstd, kmean, kstd,
      wt1, b1, wt2, b2, wt3, b3, wt4, b4p, (float*)d_out);
}
